// SelfAttention_46488726012092
// MI455X (gfx1250) — matrix operand order repaired and checked by reading, not compiled, not run on hardware
//
#include <hip/hip_runtime.h>
#include <hip/hip_bf16.h>

#define BB 8
#define LL 256
#define FF 512
#define UU 512

typedef __attribute__((ext_vector_type(16))) __bf16 v16bf;
typedef __attribute__((ext_vector_type(8)))  __bf16 v8bf;
typedef __attribute__((ext_vector_type(8)))  float  v8f;

union BF16x16 { v16bf v; v8bf h[2]; };

// ---- native tanh: builtin if declared, else raw CDNA5 v_tanh_f32 ----------
#if defined(__has_builtin)
#if __has_builtin(__builtin_amdgcn_tanhf)
#define HAVE_TANH_BUILTIN 1
#endif
#endif

static __device__ __forceinline__ float fast_tanh(float x) {
#ifdef HAVE_TANH_BUILTIN
    return __builtin_amdgcn_tanhf(x);
#else
    float r;
    asm("v_tanh_f32 %0, %1" : "=v"(r) : "v"(x));
    return r;
#endif
}

#define EXPF(x) __builtin_amdgcn_exp2f((x) * 1.4426950408889634f)

// ---- WMMA fragment loaders (ISA 7.12.2 16-bit layouts) --------------------
static __device__ __forceinline__ v16bf load_a(const __bf16* __restrict__ row,
                                               int k0, int hi) {
    BF16x16 a;
    a.h[0] = *(const v8bf*)(row + k0 + 8 * hi);       // K = 0..7   (+8 if hi)
    a.h[1] = *(const v8bf*)(row + k0 + 16 + 8 * hi);  // K = 16..23 (+8 if hi)
    return a.v;
}

static __device__ __forceinline__ v16bf load_b(const __bf16* __restrict__ row,
                                               int k0, int hi) {
    BF16x16 b;
    int kb = k0 + 8 * hi;                             // K = 0..7, 16..23 (+8 if hi)
    b.h[0] = *(const v8bf*)(row + kb);
    b.h[1] = *(const v8bf*)(row + kb + 16);
    return b.v;
}

#define WMMA_BF16(A, B, C) \
    __builtin_amdgcn_wmma_f32_16x16x32_bf16(false, (A), false, (B), (short)0, (C), false, false)

// ---------------------------------------------------------------------------
// Prep: fp32 -> bf16 copies of X (row-major and transposed per batch)
// ---------------------------------------------------------------------------
__global__ __launch_bounds__(256) void prep_x(const float* __restrict__ X,
                                              __bf16* __restrict__ Xbf,
                                              __bf16* __restrict__ Xt) {
    int i = blockIdx.x * 256 + threadIdx.x;          // over B*L*F = 1M
    int f = i & 511;
    int l = (i >> 9) & 255;
    int b = i >> 17;
    float v = X[i];
    __bf16 h = (__bf16)v;
    Xbf[i] = h;
    Xt[(b * FF + f) * LL + l] = h;                   // [b][f][l]
}

// Transpose + convert weight [F][U] -> [U][F] bf16
__global__ __launch_bounds__(256) void prep_w(const float* __restrict__ W,
                                              __bf16* __restrict__ Wt) {
    int i = blockIdx.x * 256 + threadIdx.x;          // over F*U = 256K
    int u = i & 511;
    int f = i >> 9;
    Wt[u * FF + f] = (__bf16)W[i];
}

// ---------------------------------------------------------------------------
// Projection GEMM: [2048 x 512] = Xbf * Wt^T (Wt stored [N][K]).
// One wave computes a 32x32 block (2x2 WMMA tiles): 4 fragment loads feed
// 4 v_wmma_f32_16x16x32_bf16 per K-step.
// mode 0: Out[row][u] = gemm + bias[u]       (Q, row-major)
// mode 1: OutT[b][u][l] = gemm               (K, transposed for score kernel)
// ---------------------------------------------------------------------------
__global__ __launch_bounds__(256) void proj_gemm(const __bf16* __restrict__ Xbf,
                                                 const __bf16* __restrict__ Wt,
                                                 const float* __restrict__ bias,
                                                 float* __restrict__ Out,
                                                 float* __restrict__ OutT,
                                                 int mode) {
    int wave = threadIdx.x >> 5;
    int lane = threadIdx.x & 31;
    int tile = blockIdx.x * 8 + wave;                // 1024 macro tiles = 64 x 16
    int tm = tile >> 4;                              // 0..63  (32-row block)
    int tn = tile & 15;                              // 0..15  (32-col block)
    int m0 = tm << 5, n0 = tn << 5;
    int mr = lane & 15;
    int hi = lane >> 4;

    const __bf16* arow0 = Xbf + (size_t)(m0 + mr) * FF;
    const __bf16* arow1 = arow0 + (size_t)16 * FF;
    const __bf16* brow0 = Wt  + (size_t)(n0 + mr) * FF;
    const __bf16* brow1 = brow0 + (size_t)16 * FF;

    v8f c00 = {}, c01 = {}, c10 = {}, c11 = {};
#pragma unroll
    for (int k0 = 0; k0 < FF; k0 += 32) {
        v16bf a0 = load_a(arow0, k0, hi);
        v16bf a1 = load_a(arow1, k0, hi);
        v16bf b0 = load_b(brow0, k0, hi);
        v16bf b1 = load_b(brow1, k0, hi);
        c00 = WMMA_BF16(a0, b0, c00);
        c01 = WMMA_BF16(a0, b1, c01);
        c10 = WMMA_BF16(a1, b0, c10);
        c11 = WMMA_BF16(a1, b1, c11);
    }

    if (mode == 0) {
        float bv0 = bias[n0 + mr];
        float bv1 = bias[n0 + 16 + mr];
#pragma unroll
        for (int r = 0; r < 8; ++r) {
            int row0 = m0 + 8 * hi + r;
            int row1 = row0 + 16;
            Out[(size_t)row0 * UU + n0 + mr]      = c00[r] + bv0;
            Out[(size_t)row0 * UU + n0 + 16 + mr] = c01[r] + bv1;
            Out[(size_t)row1 * UU + n0 + mr]      = c10[r] + bv0;
            Out[(size_t)row1 * UU + n0 + 16 + mr] = c11[r] + bv1;
        }
    } else {
        int b_ = m0 >> 8;                            // batch (32 | 256)
        int l0 = (m0 & 255) + 8 * hi;
        float* p00 = OutT + ((size_t)b_ * UU + n0 + mr) * LL + l0;
        float* p01 = OutT + ((size_t)b_ * UU + n0 + 16 + mr) * LL + l0;
#pragma unroll
        for (int r = 0; r < 8; ++r) {                // contiguous along l
            p00[r]      = c00[r];
            p00[r + 16] = c10[r];
            p01[r]      = c01[r];
            p01[r + 16] = c11[r];
        }
    }
}

// ---------------------------------------------------------------------------
// Fused score + exp-normalize. Block = (b,p); thread t owns q=t.
// e[q] = sum_u Wv[u]*tanh(Q[p,u] + K[q,u]) + ba  (268M native v_tanh_f32)
// ---------------------------------------------------------------------------
__global__ __launch_bounds__(256) void scores(const float* __restrict__ Q,
                                              const float* __restrict__ Kt,
                                              const float* __restrict__ Wv,
                                              const float* __restrict__ ba,
                                              __bf16* __restrict__ Abf) {
    int bp = blockIdx.x;                 // b*256 + p
    int b  = bp >> 8;
    int t  = threadIdx.x;                // q = t

    __shared__ float qrow[UU];
    __shared__ float wv[UU];
    __shared__ float red[256];

    qrow[t]       = Q[(size_t)bp * UU + t];
    qrow[t + 256] = Q[(size_t)bp * UU + t + 256];
    wv[t]         = Wv[t];
    wv[t + 256]   = Wv[t + 256];
    __syncthreads();

    const float* kcol = Kt + (size_t)b * UU * LL + t;   // Kt[b][u][t], coalesced
    float acc = 0.f;
#pragma unroll 16
    for (int u = 0; u < UU; ++u)
        acc += wv[u] * fast_tanh(qrow[u] + kcol[(size_t)u * LL]);

    float e = acc + ba[0];

    // row max
    red[t] = e; __syncthreads();
#pragma unroll
    for (int s = 128; s > 0; s >>= 1) {
        if (t < s) red[t] = fmaxf(red[t], red[t + s]);
        __syncthreads();
    }
    float m = red[0];
    __syncthreads();

    float ex = EXPF(e - m);
    red[t] = ex; __syncthreads();
#pragma unroll
    for (int s = 128; s > 0; s >>= 1) {
        if (t < s) red[t] += red[t + s];
        __syncthreads();
    }
    float a = ex / (red[0] + 1e-7f);

    Abf[(size_t)bp * LL + t] = (__bf16)a;
}

// ---------------------------------------------------------------------------
// Output GEMM per batch: out[b] = A[b] (256x256) @ X[b] (256x512), f32 accum.
// Same 2x2 WMMA blocking (32x32 per wave).
// ---------------------------------------------------------------------------
__global__ __launch_bounds__(256) void out_gemm(const __bf16* __restrict__ Abf,
                                                const __bf16* __restrict__ Xt,
                                                float* __restrict__ Out) {
    int wave = threadIdx.x >> 5;
    int lane = threadIdx.x & 31;
    int tile = blockIdx.x * 8 + wave;    // 1024 macro tiles = 8 batches * 8 * 16
    int b  = tile >> 7;
    int t2 = tile & 127;
    int tm = t2 >> 4;                    // 0..7   (32-row block of p)
    int tn = t2 & 15;                    // 0..15  (32-col block of f)
    int m0 = tm << 5, n0 = tn << 5;
    int mr = lane & 15;
    int hi = lane >> 4;

    const __bf16* arow0 = Abf + ((size_t)b * LL + m0 + mr) * LL;
    const __bf16* arow1 = arow0 + (size_t)16 * LL;
    const __bf16* brow0 = Xt  + ((size_t)b * FF + n0 + mr) * LL;
    const __bf16* brow1 = brow0 + (size_t)16 * LL;

    v8f c00 = {}, c01 = {}, c10 = {}, c11 = {};
#pragma unroll
    for (int k0 = 0; k0 < LL; k0 += 32) {
        v16bf a0 = load_a(arow0, k0, hi);
        v16bf a1 = load_a(arow1, k0, hi);
        v16bf b0 = load_b(brow0, k0, hi);
        v16bf b1 = load_b(brow1, k0, hi);
        c00 = WMMA_BF16(a0, b0, c00);
        c01 = WMMA_BF16(a0, b1, c01);
        c10 = WMMA_BF16(a1, b0, c10);
        c11 = WMMA_BF16(a1, b1, c11);
    }

#pragma unroll
    for (int r = 0; r < 8; ++r) {
        int row0 = m0 + 8 * hi + r;
        int row1 = row0 + 16;
        float* o0 = Out + ((size_t)b * LL + row0) * FF;
        float* o1 = Out + ((size_t)b * LL + row1) * FF;
        o0[n0 + mr]      = c00[r];
        o0[n0 + 16 + mr] = c01[r];
        o1[n0 + mr]      = c10[r];
        o1[n0 + 16 + mr] = c11[r];
    }
}

// ---------------------------------------------------------------------------
extern "C" void kernel_launch(void* const* d_in, const int* in_sizes, int n_in,
                              void* d_out, int out_size, void* d_ws, size_t ws_size,
                              hipStream_t stream) {
    const float* X  = (const float*)d_in[0];
    const float* Wq = (const float*)d_in[1];
    const float* Wk = (const float*)d_in[2];
    const float* Wv = (const float*)d_in[3];
    const float* bh = (const float*)d_in[4];
    const float* ba = (const float*)d_in[5];
    float* Out = (float*)d_out;

    char* ws = (char*)d_ws;
    __bf16* Xbf = (__bf16*)(ws + 0);              // 2 MB  [b][l][f]
    __bf16* Xt  = (__bf16*)(ws + (2u << 20));     // 2 MB  [b][f][l]
    __bf16* Wqt = (__bf16*)(ws + (4u << 20));     // 512 KB [u][f]
    __bf16* Wkt = (__bf16*)(ws + (4u << 20) + (512u << 10));
    float*  Q   = (float*)(ws + (5u << 20));      // 4 MB  [b][l][u] (+bh)
    float*  Kt  = (float*)(ws + (9u << 20));      // 4 MB  [b][u][l]
    __bf16* Abf = (__bf16*)(ws + (13u << 20));    // 1 MB  [b][p][q]

    prep_x<<<4096, 256, 0, stream>>>(X, Xbf, Xt);
    prep_w<<<1024, 256, 0, stream>>>(Wq, Wqt);
    prep_w<<<1024, 256, 0, stream>>>(Wk, Wkt);

    proj_gemm<<<128, 256, 0, stream>>>(Xbf, Wqt, bh, Q, nullptr, 0);
    proj_gemm<<<128, 256, 0, stream>>>(Xbf, Wkt, nullptr, nullptr, Kt, 1);

    scores<<<2048, 256, 0, stream>>>(Q, Kt, Wv, ba, Abf);

    out_gemm<<<128, 256, 0, stream>>>(Abf, Xt, Out);
}
